// HetAgg_26087631356646
// MI455X (gfx1250) — compile-verified, matrix operand order checked by reading, not run on hardware
//
#include <hip/hip_runtime.h>
#include <hip/hip_bf16.h>

typedef __attribute__((ext_vector_type(16))) _Float16 v16h;
typedef __attribute__((ext_vector_type(8)))  _Float16 v8h;
typedef __attribute__((ext_vector_type(8)))  float    v8f;

#define EMBED   128
#define HID     64
#define NGATE   512            // 4*HID * 2 directions
#define BATCH   200000
#define GENE_N  200000
#define MTILES  (BATCH / 16)   // 12500, exact

// Load a 16-half WMMA fragment: halves [0..7] from p[0..7], halves [8..15]
// from p[16..23].  Matches the CDNA5 16-bit A/B per-lane K layout when the
// base pointer is chosen per-lane (see call sites).
__device__ __forceinline__ v16h ld_frag(const _Float16* p) {
    union { v16h v; v8h h[2]; } u;
    u.h[0] = *(const v8h*)(p);
    u.h[1] = *(const v8h*)(p + 16);
    return u.v;
}

__device__ __forceinline__ float fsigmoid(float x) {
    return 1.0f / (1.0f + __expf(-x));
}
__device__ __forceinline__ float ftanh(float x) {
    float e = __expf(-2.0f * fabsf(x));
    float t = (1.0f - e) / (1.0f + e);
    return copysignf(t, x);
}

// Convert one table's (Wf | Wb) f32 weights -> f16, concatenated [512][128].
__global__ void cvtW_kernel(const float* __restrict__ Wf,
                            const float* __restrict__ Wb,
                            _Float16* __restrict__ dst) {
    int i = blockIdx.x * blockDim.x + threadIdx.x;   // 0 .. 65535
    if (i < 256 * EMBED)       dst[i] = (_Float16)Wf[i];
    else if (i < 512 * EMBED)  dst[i] = (_Float16)Wb[i - 256 * EMBED];
}

__global__ __launch_bounds__(256, 1)
void hetagg_kernel(const float* __restrict__ gene_feat,
                   const float* __restrict__ cell_feat,
                   const _Float16* __restrict__ WgH,
                   const _Float16* __restrict__ WcH,
                   const float* __restrict__ gbf, const float* __restrict__ gbb,
                   const float* __restrict__ cbf, const float* __restrict__ cbb,
                   const int* __restrict__ c_ids,
                   const int* __restrict__ p_ids,
                   const int* __restrict__ n_ids,
                   float* __restrict__ out) {
    __shared__ alignas(16) _Float16 Alds[16 * EMBED];   // 4 KB  f16 A tile
    __shared__ float Gl[16 * NGATE];                    // 32 KB gate tile
    __shared__ float Blds[NGATE];                       // 2 KB  biases

    const int tid  = threadIdx.x;
    const int lane = tid & 31;
    const int wave = tid >> 5;          // 0..7, owns N-tiles wave*4 .. wave*4+3
    const int task = blockIdx.y;        // 0=center(gene), 1=pos(cell), 2=neg(cell)

    const float* feat;  const _Float16* Wh;
    const float* bf;    const float* bb;
    const int* ids;     int idoff;
    if (task == 0)      { feat = gene_feat; Wh = WgH; bf = gbf; bb = gbb; ids = c_ids; idoff = 0; }
    else if (task == 1) { feat = cell_feat; Wh = WcH; bf = cbf; bb = cbb; ids = p_ids; idoff = GENE_N; }
    else                { feat = cell_feat; Wh = WcH; bf = cbf; bb = cbb; ids = n_ids; idoff = GENE_N; }
    float* outT = out + (size_t)task * BATCH * EMBED;

    // biases -> LDS (first barrier in the loop covers visibility)
    Blds[tid]       = bf[tid];
    Blds[256 + tid] = bb[tid];

    // ---- B fragments: register-resident for the whole block ----
    // B is Kx N (column n == row n of W, gates[m][n] = sum_k X[m][k]*W[n][k]).
    // Per-lane layout mirrors the 16-bit A layout: lanes 0-15 hold K 0-7 &
    // 16-23 of the 32-wide k-step, lanes 16-31 hold K 8-15 & 24-31.
    const int nsel = lane & 15;
    const int sel  = lane >> 4;         // 0/1 -> +0 / +8 K offset
    v16h bfrag[4][4];
#pragma unroll
    for (int t = 0; t < 4; ++t) {
        const int n = (wave * 4 + t) * 16 + nsel;
#pragma unroll
        for (int kk = 0; kk < 4; ++kk) {
            bfrag[t][kk] = ld_frag(Wh + (size_t)n * EMBED + kk * 32 + sel * 8);
        }
    }

    const int row = tid >> 4;           // 0..15 (gather / elementwise role)
    const int seg = tid & 15;           // 8 consecutive columns each

    for (int mt = blockIdx.x; mt < MTILES; mt += gridDim.x) {
        // ---- gather 16 rows, convert f32 -> f16 into LDS ----
        {
            const int gr = mt * 16 + row;
            const int id = ids[gr] - idoff;
            const float4* fp = (const float4*)(feat + (size_t)id * EMBED + seg * 8);
            float4 f0 = fp[0], f1 = fp[1];
            v8h h;
            h[0] = (_Float16)f0.x; h[1] = (_Float16)f0.y;
            h[2] = (_Float16)f0.z; h[3] = (_Float16)f0.w;
            h[4] = (_Float16)f1.x; h[5] = (_Float16)f1.y;
            h[6] = (_Float16)f1.z; h[7] = (_Float16)f1.w;
            *(v8h*)&Alds[row * EMBED + seg * 8] = h;

            // prefetch next tile's gather target (global_prefetch_b8)
            const int ngr = gr + gridDim.x * 16;
            if (ngr < BATCH) {
                const int nid = ids[ngr] - idoff;
                __builtin_prefetch(feat + (size_t)nid * EMBED + seg * 8, 0, 1);
            }
        }
        __syncthreads();

        // ---- GEMM tile: 16 x 512 over K=128 via 16 WMMAs per wave ----
        v8f acc[4] = {};
#pragma unroll
        for (int kk = 0; kk < 4; ++kk) {
            v16h a = ld_frag(&Alds[nsel * EMBED + kk * 32 + sel * 8]);
#pragma unroll
            for (int t = 0; t < 4; ++t) {
                acc[t] = __builtin_amdgcn_wmma_f32_16x16x32_f16(
                    false, a, false, bfrag[t][kk], (short)0, acc[t], false, false);
            }
        }
        // C layout: VGPR r, lane l -> M = r + 8*(l>>4), N = l&15
#pragma unroll
        for (int t = 0; t < 4; ++t) {
            const int col = (wave * 4 + t) * 16 + nsel;
#pragma unroll
            for (int r = 0; r < 8; ++r) {
                Gl[(sel * 8 + r) * NGATE + col] = acc[t][r];
            }
        }
        __syncthreads();

        // ---- LSTM nonlinearity + store (f gate irrelevant, c0 = 0) ----
        {
            const int gr = mt * 16 + row;
            float res[8];
#pragma unroll
            for (int q = 0; q < 8; ++q) {
                const int j   = seg * 8 + q;        // output column 0..127
                const int dir = j >> 6;             // 0 = fwd, 1 = bwd
                const int jj  = j & 63;
                const int gb  = dir * 256;
                float iv = Gl[row * NGATE + gb + jj]        + Blds[gb + jj];
                float gv = Gl[row * NGATE + gb + 128 + jj]  + Blds[gb + 128 + jj];
                float ov = Gl[row * NGATE + gb + 192 + jj]  + Blds[gb + 192 + jj];
                float c  = fsigmoid(iv) * ftanh(gv);
                res[q]   = fsigmoid(ov) * ftanh(c);
            }
            float4* op = (float4*)(outT + (size_t)gr * EMBED + seg * 8);
            op[0] = make_float4(res[0], res[1], res[2], res[3]);
            op[1] = make_float4(res[4], res[5], res[6], res[7]);
        }
        // No trailing barrier needed: next iteration's first __syncthreads()
        // orders this elementwise phase before the next Gl overwrite, and the
        // A-tile rewrite only races with completed WMMA reads.
    }
}

extern "C" void kernel_launch(void* const* d_in, const int* in_sizes, int n_in,
                              void* d_out, int out_size, void* d_ws, size_t ws_size,
                              hipStream_t stream) {
    (void)in_sizes; (void)n_in; (void)out_size; (void)ws_size;
    const float* gene_feat = (const float*)d_in[0];
    const float* cell_feat = (const float*)d_in[1];
    const float* gene_Wf   = (const float*)d_in[2];
    const float* gene_bf   = (const float*)d_in[3];
    const float* gene_Wb   = (const float*)d_in[4];
    const float* gene_bb   = (const float*)d_in[5];
    const float* cell_Wf   = (const float*)d_in[6];
    const float* cell_bf   = (const float*)d_in[7];
    const float* cell_Wb   = (const float*)d_in[8];
    const float* cell_bb   = (const float*)d_in[9];
    const int*   c_ids     = (const int*)d_in[10];
    const int*   p_ids     = (const int*)d_in[11];
    const int*   n_ids     = (const int*)d_in[12];
    float* out = (float*)d_out;

    _Float16* WgH = (_Float16*)d_ws;              // [512][128] f16
    _Float16* WcH = WgH + 512 * EMBED;            // [512][128] f16 (256 KB total)

    cvtW_kernel<<<256, 256, 0, stream>>>(gene_Wf, gene_Wb, WgH);
    cvtW_kernel<<<256, 256, 0, stream>>>(cell_Wf, cell_Wb, WcH);

    dim3 grid(512, 3);
    hetagg_kernel<<<grid, 256, 0, stream>>>(gene_feat, cell_feat, WgH, WcH,
                                            gene_bf, gene_bb, cell_bf, cell_bb,
                                            c_ids, p_ids, n_ids, out);
}
